// AttentionHead_67147518705757
// MI455X (gfx1250) — matrix operand order repaired and checked by reading, not compiled, not run on hardware
//
#include <hip/hip_runtime.h>
#include <cstdint>

typedef __bf16 bf16;
typedef __attribute__((ext_vector_type(8)))  __bf16 v8bf;
typedef __attribute__((ext_vector_type(16))) __bf16 v16bf;
typedef __attribute__((ext_vector_type(8)))  float  v8f;

#define DEVFN static __device__ __forceinline__

constexpr int Bn = 4;      // batch
constexpr int Cc = 256;    // channels
constexpr int QKc = 64;    // qk dim
constexpr int Nc = 4096;   // h*w

DEVFN v8f vzero(){
  v8f r;
#pragma unroll
  for(int i=0;i<8;++i) r[i] = 0.f;
  return r;
}

DEVFN v16bf cat16(v8bf lo, v8bf hi){
  v16bf r;
#pragma unroll
  for(int i=0;i<8;++i){ r[i] = lo[i]; r[i+8] = hi[i]; }
  return r;
}

DEVFN v8f wmma_bf16(v16bf a, v16bf b, v8f c){
  // D = A(16x32 bf16) * B(32x16 bf16) + C(16x16 f32)
  return __builtin_amdgcn_wmma_f32_16x16x32_bf16(false, a, false, b, (short)0, c, false, false);
}

// Async global->LDS copy of 16 bytes (CDNA5 GLOBAL_LOAD_ASYNC_TO_LDS_B128,
// tracked by ASYNCcnt).  lds_addr = LDS byte offset (low 32 bits of a generic
// shared pointer per the flat-address aperture rules).
DEVFN void async_ld_b128(uint32_t lds_addr, const void* gptr){
  asm volatile("global_load_async_to_lds_b128 %0, %1, off"
               :: "v"(lds_addr), "v"((uint64_t)(uintptr_t)gptr) : "memory");
}
DEVFN void wait_async0(){
  asm volatile("s_wait_asynccnt 0x0" ::: "memory");
}

// A fragment from row-major [M][K] storage, 16x32 tile at (row0, k0).
// CDNA5 16-bit A layout: lane = M(low 4 bits); lanes<16 hold K {k0..k0+7, k0+16..k0+23},
// lanes>=16 hold K {k0+8..k0+15, k0+24..k0+31}.  Two contiguous 16B loads per lane.
DEVFN v16bf load_a(const bf16* A, int lda, int row0, int k0, int lane){
  int m    = row0 + (lane & 15);
  int koff = (lane < 16) ? 0 : 8;
  const bf16* p = A + (size_t)m * lda + k0 + koff;
  return cat16(*(const v8bf*)p, *(const v8bf*)(p + 16));
}

// B fragment from [N][K] ("transposed") storage: Bt[n][k]. 32x16 tile at (col0, k0).
// CDNA5 16-bit B layout: N = lane&15; lanes<16 hold K {k0..k0+7, k0+16..k0+23},
// lanes>=16 hold K {k0+8..k0+15, k0+24..k0+31}.  Two 16B loads per lane.
DEVFN v16bf load_b_t(const bf16* Bt, int ldk, int col0, int k0, int lane){
  int c  = col0 + (lane & 15);
  int kb = (lane >> 4) << 3;
  const bf16* p = Bt + (size_t)c * ldk + k0 + kb;
  return cat16(*(const v8bf*)p, *(const v8bf*)(p + 16));
}

// ---------------------------------------------------------------------------
// K0: convert weights fp32 -> bf16.  Wstk rows: [0,64)=WQ, [64,128)=WK, [128,384)=WV
// ---------------------------------------------------------------------------
__global__ void cvt_weights(const float* __restrict__ WQ, const float* __restrict__ WK,
                            const float* __restrict__ WV, const float* __restrict__ W1,
                            const float* __restrict__ W2, bf16* __restrict__ Wstk,
                            bf16* __restrict__ W1b, bf16* __restrict__ W2b){
  int i = blockIdx.x * 256 + threadIdx.x;
  if      (i <  16384) Wstk[i]        = (bf16)WQ[i];
  else if (i <  32768) Wstk[i]        = (bf16)WK[i - 16384];
  else if (i <  98304) Wstk[i]        = (bf16)WV[i - 32768];
  else if (i < 163840) W1b[i -  98304] = (bf16)W1[i -  98304];
  else if (i < 229376) W2b[i - 163840] = (bf16)W2[i - 163840];
}

// ---------------------------------------------------------------------------
// K1: QKV projection.  q' = (WQ x + bQ + PE)/8 -> qT[b][n][64] (1/8 folded into q)
//     k  =  WK x + bK + PE                      -> kT[b][n][64] (transposed: [N][K])
//     v  =  WV x + bV                           -> vM[b][256][n]
// ---------------------------------------------------------------------------
__global__ __launch_bounds__(256) void proj_qkv(
    const float* __restrict__ x, const bf16* __restrict__ Wstk,
    const float* __restrict__ bQ, const float* __restrict__ bK,
    const float* __restrict__ bV, const float* __restrict__ PE,
    bf16* __restrict__ qT, bf16* __restrict__ kT, bf16* __restrict__ vM){
  __shared__ __align__(16) bf16 ldsX[64 * 256];          // [n][c] (transposed, bf16)
  const int tid = threadIdx.x, lane = tid & 31, w = tid >> 5;
  const int nt = blockIdx.x, ot = blockIdx.y, b = blockIdx.z;
  const int n0 = nt * 64;
  const float* xb = x + (size_t)b * Cc * Nc;

  for(int idx = tid; idx < 4096; idx += 256){            // 256 c-rows x 16 float4 chunks
    int c = idx >> 4, ch = (idx & 15) * 4;
    float4 f = *(const float4*)(xb + (size_t)c * Nc + n0 + ch);
    ldsX[(ch + 0) * 256 + c] = (bf16)f.x;
    ldsX[(ch + 1) * 256 + c] = (bf16)f.y;
    ldsX[(ch + 2) * 256 + c] = (bf16)f.z;
    ldsX[(ch + 3) * 256 + c] = (bf16)f.w;
  }
  __syncthreads();

  const int osub = w & 3, nb = (w >> 2) * 32;
  const bf16* A = Wstk + (size_t)ot * 64 * 256;
  v8f acc[2] = { vzero(), vzero() };
  for(int k0 = 0; k0 < 256; k0 += 32){
    v16bf a = load_a(A, 256, osub * 16, k0, lane);
#pragma unroll
    for(int ns = 0; ns < 2; ++ns)
      acc[ns] = wmma_bf16(a, load_b_t(ldsX, 256, nb + ns * 16, k0, lane), acc[ns]);
  }

  const int mb = (lane < 16) ? 0 : 8;
#pragma unroll
  for(int ns = 0; ns < 2; ++ns){
    int n = n0 + nb + ns * 16 + (lane & 15);
    if(ot == 0){                                         // Q: +bias +PE, scale 1/8, [n][o]
      int ob = osub * 16 + mb;
      v8bf pk;
#pragma unroll
      for(int r = 0; r < 8; ++r){
        int o = ob + r;
        pk[r] = (bf16)((acc[ns][r] + bQ[o] + PE[(size_t)o * Nc + n]) * 0.125f);
      }
      *(v8bf*)(qT + ((size_t)b * Nc + n) * QKc + ob) = pk;
    } else if(ot == 1){                                  // K: +bias +PE, transposed [n][o]
      int ob = osub * 16 + mb;
      v8bf pk;
#pragma unroll
      for(int r = 0; r < 8; ++r){
        int o = ob + r;
        pk[r] = (bf16)(acc[ns][r] + bK[o] + PE[(size_t)o * Nc + n]);
      }
      *(v8bf*)(kT + ((size_t)b * Nc + n) * QKc + ob) = pk;
    } else {                                             // V: +bias, [c][n]
#pragma unroll
      for(int r = 0; r < 8; ++r){
        int o = (ot - 2) * 64 + osub * 16 + mb + r;
        vM[((size_t)b * Cc + o) * Nc + n] = (bf16)(acc[ns][r] + bV[o]);
      }
    }
  }
}

// ---------------------------------------------------------------------------
// K2: per-KEY-column softmax stats (softmax over query axis i).
// Each block owns 64 columns j; streams all i tiles with WMMA keeping online
// (max, sumexp) per column in registers.  Outputs m[b][j] and 1/d[b][j].
// ---------------------------------------------------------------------------
__global__ __launch_bounds__(256) void col_softmax_stats(
    const bf16* __restrict__ qT, const bf16* __restrict__ kT,
    float* __restrict__ mOut, float* __restrict__ rdOut){
  __shared__ float lm[8][64];
  __shared__ float ls[8][64];
  const int tid = threadIdx.x, lane = tid & 31, w = tid >> 5;
  const int jt = blockIdx.x, b = blockIdx.y;
  const int j0 = jt * 64;
  const bf16* q  = qT + (size_t)b * Nc * QKc;
  const bf16* kt = kT + (size_t)b * Nc * QKc;

  v16bf bk[4][2];                                        // K fragments fixed per block
#pragma unroll
  for(int js = 0; js < 4; ++js){
    bk[js][0] = load_b_t(kt, QKc, j0 + js * 16, 0,  lane);
    bk[js][1] = load_b_t(kt, QKc, j0 + js * 16, 32, lane);
  }
  float rm[4], rs[4];
#pragma unroll
  for(int js = 0; js < 4; ++js){ rm[js] = -3.0e38f; rs[js] = 0.f; }

  for(int i0 = 0; i0 < Nc; i0 += 128){                   // 8 waves cover 128 rows / iter
    v16bf a0 = load_a(q, QKc, i0 + w * 16, 0,  lane);
    v16bf a1 = load_a(q, QKc, i0 + w * 16, 32, lane);
#pragma unroll
    for(int js = 0; js < 4; ++js){
      v8f s = vzero();
      s = wmma_bf16(a0, bk[js][0], s);
      s = wmma_bf16(a1, bk[js][1], s);
      float vmax = s[0];
#pragma unroll
      for(int r = 1; r < 8; ++r) vmax = fmaxf(vmax, s[r]);
      float mn  = fmaxf(rm[js], vmax);
      float acc = rs[js] * __expf(rm[js] - mn);
#pragma unroll
      for(int r = 0; r < 8; ++r) acc += __expf(s[r] - mn);
      rm[js] = mn; rs[js] = acc;
    }
  }
  // combine the two lane-halves that own each column, then combine 8 waves via LDS
#pragma unroll
  for(int js = 0; js < 4; ++js){
    float m2 = __shfl_xor(rm[js], 16);
    float s2 = __shfl_xor(rs[js], 16);
    float mf = fmaxf(rm[js], m2);
    float sf = rs[js] * __expf(rm[js] - mf) + s2 * __expf(m2 - mf);
    if(lane < 16){ lm[w][js * 16 + lane] = mf; ls[w][js * 16 + lane] = sf; }
  }
  __syncthreads();
  if(tid < 64){
    float mf = -3.0e38f;
#pragma unroll
    for(int ww = 0; ww < 8; ++ww) mf = fmaxf(mf, lm[ww][tid]);
    float sf = 0.f;
#pragma unroll
    for(int ww = 0; ww < 8; ++ww) sf += ls[ww][tid] * __expf(lm[ww][tid] - mf);
    mOut [(size_t)b * Nc + j0 + tid] = mf;
    rdOut[(size_t)b * Nc + j0 + tid] = 1.0f / sf;
  }
}

// ---------------------------------------------------------------------------
// K4: O[c][i] = sum_j exp(S[i][j]-m_j)*rd_j * v[j][c].  S recomputed on the fly;
// exp tile routed through LDS (C-layout -> A-layout).  V tile is copied with
// async global->LDS DMA, overlapping the S-tile WMMAs of the same iteration.
// ---------------------------------------------------------------------------
__global__ __launch_bounds__(256) void attn_pv(
    const bf16* __restrict__ qT, const bf16* __restrict__ kT,
    const bf16* __restrict__ vM, const float* __restrict__ mArr,
    const float* __restrict__ rdArr, bf16* __restrict__ Ob){
  __shared__ __align__(16) bf16 ldsE[64 * 64];           // [i][j]
  __shared__ __align__(16) bf16 ldsV[256 * 64];          // [c][j]  ([N][K] layout)
  const int tid = threadIdx.x, lane = tid & 31, w = tid >> 5;
  const int it = blockIdx.x, b = blockIdx.y;
  const int i0 = it * 64;
  const bf16* q  = qT + (size_t)b * Nc * QKc;
  const bf16* kt = kT + (size_t)b * Nc * QKc;
  const bf16* vm = vM + (size_t)b * Cc * Nc;
  const float* mj = mArr  + (size_t)b * Nc;
  const float* rj = rdArr + (size_t)b * Nc;

  const int isub = w & 3, half = w >> 2;
  const int jb = half * 32, cb = half * 128;
  const int mb = (lane < 16) ? 0 : 8;
  const uint32_t ldsv_base = (uint32_t)(uintptr_t)(&ldsV[0]);

  v16bf a0 = load_a(q, QKc, i0 + isub * 16, 0,  lane);   // Q fragments fixed per block
  v16bf a1 = load_a(q, QKc, i0 + isub * 16, 32, lane);
  v8f acc[8];
#pragma unroll
  for(int cs = 0; cs < 8; ++cs) acc[cs] = vzero();

  for(int j0 = 0; j0 < Nc; j0 += 64){
    // kick off async DMA of the V tile into LDS (overlaps with S WMMAs below)
    for(int idx = tid; idx < 2048; idx += 256){
      int c = idx >> 3, ch = (idx & 7) * 8;
      async_ld_b128(ldsv_base + (uint32_t)(c * 64 + ch) * 2u,
                    vm + (size_t)c * Nc + j0 + ch);
    }
    // S = q' . k for this wave's two j-subtiles, then E = exp(s-m_j)*rd_j into LDS
#pragma unroll
    for(int js = 0; js < 2; ++js){
      v8f s = vzero();
      s = wmma_bf16(a0, load_b_t(kt, QKc, j0 + jb + js * 16, 0,  lane), s);
      s = wmma_bf16(a1, load_b_t(kt, QKc, j0 + jb + js * 16, 32, lane), s);
      int j = j0 + jb + js * 16 + (lane & 15);
      float mm = mj[j], rr = rj[j];
#pragma unroll
      for(int r = 0; r < 8; ++r){
        float e = __expf(s[r] - mm) * rr;
        ldsE[(isub * 16 + mb + r) * 64 + (jb + js * 16 + (lane & 15))] = (bf16)e;
      }
    }
    wait_async0();                                       // V tile landed in LDS
    __syncthreads();
    // O += E x V
#pragma unroll
    for(int ks = 0; ks < 2; ++ks){
      v16bf ae = load_a(ldsE, 64, isub * 16, ks * 32, lane);
#pragma unroll
      for(int cs = 0; cs < 8; ++cs)
        acc[cs] = wmma_bf16(ae, load_b_t(ldsV, 64, cb + cs * 16, ks * 32, lane), acc[cs]);
    }
    __syncthreads();
  }
  // store O as [c][i]: per lane 8 consecutive i -> one 16B store
#pragma unroll
  for(int cs = 0; cs < 8; ++cs){
    int c = cb + cs * 16 + (lane & 15);
    v8bf pk;
#pragma unroll
    for(int r = 0; r < 8; ++r) pk[r] = (bf16)acc[cs][r];
    *(v8bf*)(Ob + ((size_t)b * Cc + c) * Nc + i0 + isub * 16 + mb) = pk;
  }
}

// ---------------------------------------------------------------------------
// K5: hdn[n][o] = mish(W1 . O + b1)
// ---------------------------------------------------------------------------
__global__ __launch_bounds__(256) void mlp_mish(
    const bf16* __restrict__ Ob, const bf16* __restrict__ W1b,
    const float* __restrict__ b1, bf16* __restrict__ Hb){
  __shared__ __align__(16) bf16 ldsO[64 * 256];          // [n][c]
  const int tid = threadIdx.x, lane = tid & 31, w = tid >> 5;
  const int nt = blockIdx.x, ot = blockIdx.y, b = blockIdx.z;
  const int n0 = nt * 64;
  const bf16* ob = Ob + (size_t)b * Cc * Nc;
  for(int idx = tid; idx < 2048; idx += 256){            // transpose-stage [c][n]->[n][c]
    int c = idx >> 3, ch = (idx & 7) * 8;
    v8bf t = *(const v8bf*)(ob + (size_t)c * Nc + n0 + ch);
#pragma unroll
    for(int e = 0; e < 8; ++e) ldsO[(ch + e) * 256 + c] = t[e];
  }
  __syncthreads();
  const int osub = w & 3, nb = (w >> 2) * 32;
  const bf16* A = W1b + (size_t)ot * 64 * 256;
  v8f acc[2] = { vzero(), vzero() };
  for(int k0 = 0; k0 < 256; k0 += 32){
    v16bf a = load_a(A, 256, osub * 16, k0, lane);
#pragma unroll
    for(int ns = 0; ns < 2; ++ns)
      acc[ns] = wmma_bf16(a, load_b_t(ldsO, 256, nb + ns * 16, k0, lane), acc[ns]);
  }
  const int mb = (lane < 16) ? 0 : 8;
#pragma unroll
  for(int ns = 0; ns < 2; ++ns){
    int n = n0 + nb + ns * 16 + (lane & 15);
    int obase = ot * 64 + osub * 16 + mb;
    v8bf pk;
#pragma unroll
    for(int r = 0; r < 8; ++r){
      float vH = acc[ns][r] + b1[obase + r];
      float sp = log1pf(__expf(vH));                     // softplus
      pk[r] = (bf16)(vH * tanhf(sp));                    // mish
    }
    *(v8bf*)(Hb + ((size_t)b * Nc + n) * Cc + obase) = pk;
  }
}

// ---------------------------------------------------------------------------
// K6: out = W2 . hdn + b2 + x   (fp32 output); hidden tile staged via async DMA
// ---------------------------------------------------------------------------
__global__ __launch_bounds__(256) void mlp_out(
    const bf16* __restrict__ Hb, const bf16* __restrict__ W2b,
    const float* __restrict__ b2, const float* __restrict__ x,
    float* __restrict__ out){
  __shared__ __align__(16) bf16 ldsH[64 * 256];          // [n][c]
  const int tid = threadIdx.x, lane = tid & 31, w = tid >> 5;
  const int nt = blockIdx.x, ot = blockIdx.y, b = blockIdx.z;
  const int n0 = nt * 64;
  const bf16* hb = Hb + (size_t)b * Nc * Cc;
  const uint32_t ldsh_base = (uint32_t)(uintptr_t)(&ldsH[0]);
  for(int idx = tid; idx < 2048; idx += 256){            // straight copy: Hb is [n][c]
    int n = idx >> 5, ch = (idx & 31) * 8;
    async_ld_b128(ldsh_base + (uint32_t)(n * 256 + ch) * 2u,
                  hb + (size_t)(n0 + n) * Cc + ch);
  }
  wait_async0();
  __syncthreads();
  const int osub = w & 3, nb = (w >> 2) * 32;
  const bf16* A = W2b + (size_t)ot * 64 * 256;
  v8f acc[2] = { vzero(), vzero() };
  for(int k0 = 0; k0 < 256; k0 += 32){
    v16bf a = load_a(A, 256, osub * 16, k0, lane);
#pragma unroll
    for(int ns = 0; ns < 2; ++ns)
      acc[ns] = wmma_bf16(a, load_b_t(ldsH, 256, nb + ns * 16, k0, lane), acc[ns]);
  }
  const int mb = (lane < 16) ? 0 : 8;
#pragma unroll
  for(int ns = 0; ns < 2; ++ns){
    int n = n0 + nb + ns * 16 + (lane & 15);
#pragma unroll
    for(int r = 0; r < 8; ++r){
      int o = ot * 64 + osub * 16 + mb + r;
      size_t idx = ((size_t)b * Cc + o) * Nc + n;
      out[idx] = acc[ns][r] + b2[o] + x[idx];
    }
  }
}

// ---------------------------------------------------------------------------
extern "C" void kernel_launch(void* const* d_in, const int* in_sizes, int n_in,
                              void* d_out, int out_size, void* d_ws, size_t ws_size,
                              hipStream_t stream){
  (void)in_sizes; (void)n_in; (void)out_size; (void)ws_size;
  const float* x  = (const float*)d_in[0];
  const float* WQ = (const float*)d_in[1];
  const float* bQ = (const float*)d_in[2];
  const float* WK = (const float*)d_in[3];
  const float* bK = (const float*)d_in[4];
  const float* WV = (const float*)d_in[5];
  const float* bV = (const float*)d_in[6];
  const float* PE = (const float*)d_in[7];
  const float* W1 = (const float*)d_in[8];
  const float* b1 = (const float*)d_in[9];
  const float* W2 = (const float*)d_in[10];
  const float* b2 = (const float*)d_in[11];
  float* out = (float*)d_out;

  char* ws = (char*)d_ws;
  size_t off = 0;
  auto take = [&](size_t bytes) -> char* {
    char* p = ws + off;
    off += (bytes + 255) & ~(size_t)255;
    return p;
  };
  bf16*  qT    = (bf16*) take((size_t)Bn * Nc * QKc * 2);   // [b][n][64]
  bf16*  kT    = (bf16*) take((size_t)Bn * Nc * QKc * 2);   // [b][n][64]
  bf16*  vMat  = (bf16*) take((size_t)Bn * Cc * Nc * 2);    // [b][c][n]
  bf16*  Ob    = (bf16*) take((size_t)Bn * Cc * Nc * 2);    // [b][c][n]
  bf16*  Hb    = (bf16*) take((size_t)Bn * Nc * Cc * 2);    // [b][n][c]
  float* mArr  = (float*)take((size_t)Bn * Nc * 4);
  float* rdArr = (float*)take((size_t)Bn * Nc * 4);
  bf16*  Wstk  = (bf16*) take((size_t)384 * 256 * 2);
  bf16*  W1b   = (bf16*) take((size_t)256 * 256 * 2);
  bf16*  W2b   = (bf16*) take((size_t)256 * 256 * 2);

  cvt_weights<<<896, 256, 0, stream>>>(WQ, WK, WV, W1, W2, Wstk, W1b, W2b);
  proj_qkv<<<dim3(64, 6, 4), 256, 0, stream>>>(x, Wstk, bQ, bK, bV, PE, qT, kT, vMat);
  col_softmax_stats<<<dim3(64, 4), 256, 0, stream>>>(qT, kT, mArr, rdArr);
  attn_pv<<<dim3(64, 4), 256, 0, stream>>>(qT, kT, vMat, mArr, rdArr, Ob);
  mlp_mish<<<dim3(64, 4, 4), 256, 0, stream>>>(Ob, W1b, b1, Hb);
  mlp_out<<<dim3(64, 4, 4), 256, 0, stream>>>(Hb, W2b, b2, x, out);
}
